// CMRGCN_79491254714463
// MI455X (gfx1250) — matrix operand order repaired and checked by reading, not compiled, not run on hardware
//
#include <hip/hip_runtime.h>
#include <hip/hip_bf16.h>
#include <math.h>

// ---------------------------------------------------------------------------
// CMRGCN forward for MI455X (gfx1250, wave32, WMMA).
// Matmuls run on v_wmma_f32_16x16x32_f16 (f16 operands, f32 accumulate).
// Aggregation kernel stages its shared 64KB B-panel in LDS (padded, bank-
// conflict-free) and software-pipelines the A-tile global loads.
// ---------------------------------------------------------------------------

typedef __attribute__((ext_vector_type(16))) _Float16 v16h;
typedef __attribute__((ext_vector_type(8)))  _Float16 v8h;
typedef __attribute__((ext_vector_type(8)))  float    v8f;

#define Bsz     8
#define Tsz     12
#define Nsz     500
#define Dsz     64
#define NPAD    512
#define BT      (Bsz * Tsz)          // 96
#define NGr     2
#define NMIX    2
#define NLAY    2
#define NREL    8
#define Ksz     20
#define CCH     (Dsz * (NLAY + 1))   // 192 channels per mix in g
#define GSZ     ((size_t)Bsz * CCH * Nsz * Tsz)   // floats per mix in g
#define OUTPERI ((size_t)Bsz * (CCH * NGr) * Nsz * Tsz)
#define LDSPITCH 520                 // halves; 1040B = 260 dwords, 260%64=4 -> conflict-free

__device__ __forceinline__ v8f zero_v8f() {
    v8f z;
#pragma unroll
    for (int e = 0; e < 8; ++e) z[e] = 0.0f;
    return z;
}

// A operand (16x32 f16): mat is row-major [rows][ld], per-lane:
//   row = row0 + lane%16 ; 16 contiguous k at kb = k0 + 16*(lane/16)
__device__ __forceinline__ v16h load_a16x32(const _Float16* __restrict__ mat, int ld,
                                            int row0, int k0, int lane) {
    int m  = row0 + (lane & 15);
    int kb = k0 + ((lane >> 4) << 4);
    const _Float16* p = mat + (size_t)m * ld + kb;
    v8h lo = *reinterpret_cast<const v8h*>(p);
    v8h hi = *reinterpret_cast<const v8h*>(p + 8);
    v16h a;
#pragma unroll
    for (int e = 0; e < 8; ++e) { a[e] = lo[e]; a[e + 8] = hi[e]; }
    return a;
}

// B operand (32x16 f16): matT is column-major [cols][ld] (k contiguous), per-lane:
//   col = col0 + lane%16 ; 16 contiguous k at kb = k0 + 16*(lane/16)
__device__ __forceinline__ v16h load_b32x16(const _Float16* __restrict__ matT, int ld,
                                            int col0, int k0, int lane) {
    int n  = col0 + (lane & 15);
    int kb = k0 + ((lane >> 4) << 4);
    return *reinterpret_cast<const v16h*>(matT + (size_t)n * ld + kb);
}

// B operand fetched from LDS (padded pitch)
__device__ __forceinline__ v16h lds_b32x16(const _Float16* sm, int col0, int k0, int lane) {
    int n  = col0 + (lane & 15);
    int kb = k0 + ((lane >> 4) << 4);
    return *reinterpret_cast<const v16h*>(sm + (size_t)n * LDSPITCH + kb);
}

// ---------------------------------------------------------------------------
// 1) Fuse head-mixture weights: wcB/wdB stored column-major [l][rel][dout][din] f16
// ---------------------------------------------------------------------------
__global__ void fuse_w_kernel(const float* __restrict__ a_w, const float* __restrict__ B_w,
                              _Float16* __restrict__ wcB, _Float16* __restrict__ wdB) {
    int idx = blockIdx.x * blockDim.x + threadIdx.x;      // 2*8*64*64 = 65536
    if (idx >= NLAY * NREL * Dsz * Dsz) return;
    int din  = idx & 63;
    int dout = (idx >> 6) & 63;
    int rel  = (idx >> 12) & 7;
    int l    = idx >> 15;
    float sc = 0.0f, sd = 0.0f;
#pragma unroll
    for (int hd = 0; hd < 4; ++hd) {
        // B_weight [1,4,2,64,64]: [hd][l][din][dout]
        float bw = B_w[((((size_t)hd * NLAY) + l) * Dsz + din) * Dsz + dout];
        // a_weight [2,8,4,2,1,1]
        sc += a_w[(((size_t)0 * NREL + rel) * 4 + hd) * NLAY + l] * bw;
        sd += a_w[(((size_t)1 * NREL + rel) * 4 + hd) * NLAY + l] * bw;
    }
    size_t o = ((((size_t)l * NREL + rel) * Dsz + dout)) * Dsz + din;
    wcB[o] = (_Float16)sc;
    wdB[o] = (_Float16)sd;
}

__global__ void fuse_b_kernel(const float* __restrict__ a_b, const float* __restrict__ B_b,
                              float* __restrict__ bc, float* __restrict__ bd) {
    int idx = blockIdx.x * blockDim.x + threadIdx.x;      // 2*8*64 = 1024
    if (idx >= NLAY * NREL * Dsz) return;
    int d   = idx & 63;
    int rel = (idx >> 6) & 7;
    int l   = idx >> 9;
    float sc = 0.0f, sd = 0.0f;
#pragma unroll
    for (int hd = 0; hd < 4; ++hd) {
        float bb = B_b[((size_t)hd * NLAY + l) * Dsz + d];   // B_bias [1,4,2,64]
        sc += a_b[(((size_t)0 * NREL + rel) * 4 + hd) * NLAY + l] * bb;  // a_bias [2,8,4,2,1]
        sd += a_b[(((size_t)1 * NREL + rel) * 4 + hd) * NLAY + l] * bb;
    }
    bc[((size_t)l * NREL + rel) * Dsz + d] = sc;
    bd[((size_t)l * NREL + rel) * Dsz + d] = sd;
}

// ---------------------------------------------------------------------------
// 2) adjT[t][m][n] = graphs[t][n][m]  (f16, zero-padded to 512x512)
// ---------------------------------------------------------------------------
__global__ void adj_pack_kernel(const float* __restrict__ graphs, _Float16* __restrict__ adjT) {
    int idx = blockIdx.x * blockDim.x + threadIdx.x;      // 2*512*512
    if (idx >= NGr * NPAD * NPAD) return;
    int n = idx & 511;
    int m = (idx >> 9) & 511;
    int t = idx >> 18;
    float v = (m < Nsz && n < Nsz) ? graphs[((size_t)t * Nsz + n) * Nsz + m] : 0.0f;
    adjT[idx] = (_Float16)v;
}

// ---------------------------------------------------------------------------
// 3) initial pack: x[i] (B,D,N,T fp32) -> xsT f16 [i][bt][d][n(512)]  and
//    snapshot channels 0..63 of g[i] (B,192,N,T fp32)
// ---------------------------------------------------------------------------
__global__ void init_pack_kernel(const float* __restrict__ x0, const float* __restrict__ x1,
                                 _Float16* __restrict__ xsT, float* __restrict__ g) {
    size_t idx = (size_t)blockIdx.x * blockDim.x + threadIdx.x;   // 2*8*64*512*12
    if (idx >= (size_t)NMIX * Bsz * Dsz * NPAD * Tsz) return;
    int t = (int)(idx % Tsz);
    int n = (int)((idx / Tsz) % NPAD);
    int d = (int)((idx / ((size_t)Tsz * NPAD)) % Dsz);
    int b = (int)((idx / ((size_t)Tsz * NPAD * Dsz)) % Bsz);
    int i = (int)(idx / ((size_t)Tsz * NPAD * Dsz * Bsz));
    const float* x = (i == 0) ? x0 : x1;
    float v = 0.0f;
    if (n < Nsz) v = x[(((size_t)b * Dsz + d) * Nsz + n) * Tsz + t];
    int bt = b * Tsz + t;
    xsT[(((size_t)i * BT + bt) * Dsz + d) * NPAD + n] = (_Float16)v;
    if (n < Nsz)
        g[(size_t)i * GSZ + (((size_t)b * CCH + d) * Nsz + n) * Tsz + t] = v;
}

// ---------------------------------------------------------------------------
// 4) Graph aggregation: agg[combo][bt][m][d] = sum_n adjT[tg][m][n] * xsT[j][bt][d][n]
//    combo = tg*2 + j. One wave -> 16(m) x 64(d) tile, K=512 in 16 WMMA steps.
//    All 8 waves of a block share (combo, bt): the 64KB B-panel is staged in
//    LDS once per block; A tiles are software-pipelined from global.
// ---------------------------------------------------------------------------
__global__ void __launch_bounds__(256) agg_kernel(const _Float16* __restrict__ adjT,
                                                  const _Float16* __restrict__ xsT,
                                                  _Float16* __restrict__ agg) {
    __shared__ _Float16 smem[Dsz * LDSPITCH];   // 64 * 520 halves = 66,560 B

    int wid   = blockIdx.x * (blockDim.x >> 5) + (threadIdx.x >> 5);
    int lane  = threadIdx.x & 31;
    int mtile = wid & 31;                 // 32 m-tiles
    int bt    = (wid >> 5) % BT;          // 96  (uniform across the block)
    int combo = wid / (32 * BT);          // 4 combos (uniform across the block)
    int tg = combo >> 1, j = combo & 1;

    const _Float16* A  = adjT + (size_t)tg * NPAD * NPAD;
    const _Float16* Bm = xsT + ((size_t)j * BT + bt) * Dsz * NPAD;

    // Cooperative stage: 64 rows x 512 halves -> padded LDS pitch 520.
    // 4096 16-byte chunks / 256 threads = 16 per thread.
    for (int c = threadIdx.x; c < Dsz * (NPAD / 8); c += 256) {
        int row = c >> 6;              // 0..63
        int col = (c & 63) << 3;       // halves, 0..504 step 8
        *reinterpret_cast<v8h*>(smem + (size_t)row * LDSPITCH + col) =
            *reinterpret_cast<const v8h*>(Bm + (size_t)row * NPAD + col);
    }
    __syncthreads();

    v8f acc[4];
#pragma unroll
    for (int dt = 0; dt < 4; ++dt) acc[dt] = zero_v8f();

    v16h a = load_a16x32(A, NPAD, mtile * 16, 0, lane);
    for (int k = 0; k < NPAD; k += 32) {
        v16h anext = a;
        if (k + 32 < NPAD) anext = load_a16x32(A, NPAD, mtile * 16, k + 32, lane);
#pragma unroll
        for (int dt = 0; dt < 4; ++dt) {
            v16h bv = lds_b32x16(smem, dt * 16, k, lane);
            acc[dt] = __builtin_amdgcn_wmma_f32_16x16x32_f16(
                false, a, false, bv, (short)0, acc[dt], false, false);
        }
        a = anext;
    }

    _Float16* outp = agg + ((size_t)combo * BT + bt) * NPAD * Dsz;
    int mrow = mtile * 16 + ((lane >> 4) << 3);
    int dl   = lane & 15;
#pragma unroll
    for (int dt = 0; dt < 4; ++dt)
#pragma unroll
        for (int r = 0; r < 8; ++r)
            outp[(size_t)(mrow + r) * Dsz + dt * 16 + dl] = (_Float16)acc[dt][r];
}

// ---------------------------------------------------------------------------
// 5) Dense mixing: h[i][bt][m][d] = sum_{tg,j} relu(agg[tg][j]@wc + bc)
//                                         + tanh((agg[tg][j]-agg[tg][i])@wd + bd)
//    One wave -> 16(m) x 64(d) tile of h[i]. Weight panels (8KB each) are
//    shared by all waves and live in WGP$/L2.
// ---------------------------------------------------------------------------
__global__ void __launch_bounds__(256) dense_kernel(const _Float16* __restrict__ agg,
                                                    const _Float16* __restrict__ wcB,
                                                    const _Float16* __restrict__ wdB,
                                                    const float* __restrict__ bc,
                                                    const float* __restrict__ bd,
                                                    float* __restrict__ hbuf, int layer) {
    int wid   = blockIdx.x * (blockDim.x >> 5) + (threadIdx.x >> 5);
    int lane  = threadIdx.x & 31;
    int mtile = wid & 31;
    int bt    = (wid >> 5) % BT;
    int i     = wid / (32 * BT);          // mix index 0..1

    v8f hacc[4];
#pragma unroll
    for (int dt = 0; dt < 4; ++dt) hacc[dt] = zero_v8f();

    for (int cj = 0; cj < 4; ++cj) {
        int tg = cj >> 1, j = cj & 1;
        int rel = (tg * NMIX + i) * NMIX + j;
        const _Float16* Aj = agg + ((size_t)(tg * 2 + j) * BT + bt) * NPAD * Dsz;
        const _Float16* Ai = agg + ((size_t)(tg * 2 + i) * BT + bt) * NPAD * Dsz;
        const _Float16* Wc = wcB + ((size_t)layer * NREL + rel) * Dsz * Dsz;
        const _Float16* Wd = wdB + ((size_t)layer * NREL + rel) * Dsz * Dsz;

        v8f cacc[4], dacc[4];
#pragma unroll
        for (int dt = 0; dt < 4; ++dt) { cacc[dt] = zero_v8f(); dacc[dt] = zero_v8f(); }

#pragma unroll
        for (int k2 = 0; k2 < 2; ++k2) {
            v16h aj = load_a16x32(Aj, Dsz, mtile * 16, k2 * 32, lane);
            v16h ai = load_a16x32(Ai, Dsz, mtile * 16, k2 * 32, lane);
            v16h ad = aj - ai;
#pragma unroll
            for (int dt = 0; dt < 4; ++dt) {
                v16h bcv = load_b32x16(Wc, Dsz, dt * 16, k2 * 32, lane);
                v16h bdv = load_b32x16(Wd, Dsz, dt * 16, k2 * 32, lane);
                cacc[dt] = __builtin_amdgcn_wmma_f32_16x16x32_f16(
                    false, aj, false, bcv, (short)0, cacc[dt], false, false);
                dacc[dt] = __builtin_amdgcn_wmma_f32_16x16x32_f16(
                    false, ad, false, bdv, (short)0, dacc[dt], false, false);
            }
        }

#pragma unroll
        for (int dt = 0; dt < 4; ++dt) {
            int d = dt * 16 + (lane & 15);
            float bcs = bc[((size_t)layer * NREL + rel) * Dsz + d];
            float bds = bd[((size_t)layer * NREL + rel) * Dsz + d];
#pragma unroll
            for (int r = 0; r < 8; ++r) {
                float cv = cacc[dt][r] + bcs;
                cv = cv > 0.0f ? cv : 0.0f;
                float dv = tanhf(dacc[dt][r] + bds);
                hacc[dt][r] += cv + dv;
            }
        }
    }

    float* outp = hbuf + ((size_t)i * BT + bt) * NPAD * Dsz;
    int mrow = mtile * 16 + ((lane >> 4) << 3);
    int dl   = lane & 15;
#pragma unroll
    for (int dt = 0; dt < 4; ++dt)
#pragma unroll
        for (int r = 0; r < 8; ++r)
            outp[(size_t)(mrow + r) * Dsz + dt * 16 + dl] = hacc[dt][r];
}

// ---------------------------------------------------------------------------
// 6) layer pack: h (fp32, [i][bt][m][d]) -> next xsT (f16, [i][bt][d][n]) and
//    snapshot channels (layer+1)*64.. of g[i]
// ---------------------------------------------------------------------------
__global__ void layer_pack_kernel(const float* __restrict__ hbuf, _Float16* __restrict__ xsT,
                                  float* __restrict__ g, int layer) {
    size_t idx = (size_t)blockIdx.x * blockDim.x + threadIdx.x;   // 2*96*64*512
    if (idx >= (size_t)NMIX * BT * Dsz * NPAD) return;
    int n  = (int)(idx & 511);
    int d  = (int)((idx >> 9) & 63);
    int bt = (int)((idx >> 15) % BT);
    int i  = (int)(idx / ((size_t)NPAD * Dsz * BT));
    float v = 0.0f;
    if (n < Nsz) v = hbuf[(((size_t)i * BT + bt) * NPAD + n) * Dsz + d];
    xsT[(((size_t)i * BT + bt) * Dsz + d) * NPAD + n] = (_Float16)v;
    if (n < Nsz) {
        int b = bt / Tsz, t = bt % Tsz;
        g[(size_t)i * GSZ + (((size_t)b * CCH + (layer + 1) * Dsz + d) * Nsz + n) * Tsz + t] = v;
    }
}

// ---------------------------------------------------------------------------
// 7) summarize: out[i][b][tg*192+c][n][t] = sum_k g[i][b][c][nb[tg][n][k]][t]*w[tg][n][k]
// ---------------------------------------------------------------------------
__global__ void summarize_kernel(const float* __restrict__ g, const int* __restrict__ nb,
                                 const float* __restrict__ nw, float* __restrict__ out) {
    size_t idx = (size_t)blockIdx.x * blockDim.x + threadIdx.x;   // 2 * 8*384*500*12
    if (idx >= (size_t)NMIX * OUTPERI) return;
    int t  = (int)(idx % Tsz);
    int n  = (int)((idx / Tsz) % Nsz);
    int ch = (int)((idx / ((size_t)Tsz * Nsz)) % (CCH * NGr));
    int b  = (int)((idx / ((size_t)Tsz * Nsz * CCH * NGr)) % Bsz);
    int i  = (int)(idx / ((size_t)Tsz * Nsz * CCH * NGr * Bsz));
    int tg = ch / CCH;
    int c  = ch % CCH;
    const float* gp = g + (size_t)i * GSZ + (((size_t)b * CCH + c) * Nsz) * Tsz + t;
    const int*   ip = nb + ((size_t)tg * Nsz + n) * Ksz;
    const float* wp = nw + ((size_t)tg * Nsz + n) * Ksz;
    float acc = 0.0f;
#pragma unroll
    for (int k = 0; k < Ksz; ++k)
        acc += gp[(size_t)ip[k] * Tsz] * wp[k];
    out[idx] = acc;
}

// ---------------------------------------------------------------------------
extern "C" void kernel_launch(void* const* d_in, const int* in_sizes, int n_in,
                              void* d_out, int out_size, void* d_ws, size_t ws_size,
                              hipStream_t stream) {
    const float* x0     = (const float*)d_in[0];
    const float* x1     = (const float*)d_in[1];
    const float* graphs = (const float*)d_in[2];
    const int*   nbr    = (const int*)d_in[3];
    const float* nbrw   = (const float*)d_in[4];
    const float* a_w    = (const float*)d_in[5];
    const float* B_w    = (const float*)d_in[6];
    const float* a_b    = (const float*)d_in[7];
    const float* B_b    = (const float*)d_in[8];
    float* out = (float*)d_out;

    char* ws = (char*)d_ws;
    size_t off = 0;
    auto carve = [&](size_t bytes) -> void* {
        void* p = ws + off;
        off = (off + bytes + 255) & ~(size_t)255;
        return p;
    };

    _Float16* adjT = (_Float16*)carve((size_t)NGr * NPAD * NPAD * 2);
    _Float16* wcB  = (_Float16*)carve((size_t)NLAY * NREL * Dsz * Dsz * 2);
    _Float16* wdB  = (_Float16*)carve((size_t)NLAY * NREL * Dsz * Dsz * 2);
    float*    bc   = (float*)carve((size_t)NLAY * NREL * Dsz * 4);
    float*    bd   = (float*)carve((size_t)NLAY * NREL * Dsz * 4);
    _Float16* xsT  = (_Float16*)carve((size_t)NMIX * BT * Dsz * NPAD * 2);
    _Float16* agg  = (_Float16*)carve((size_t)4 * BT * NPAD * Dsz * 2);
    float*    hbuf = (float*)carve((size_t)NMIX * BT * NPAD * Dsz * 4);
    float*    gbuf = (float*)carve((size_t)NMIX * GSZ * 4);
    (void)ws_size; (void)in_sizes; (void)n_in; (void)out_size;

    fuse_w_kernel<<<(NLAY * NREL * Dsz * Dsz) / 256, 256, 0, stream>>>(a_w, B_w, wcB, wdB);
    fuse_b_kernel<<<(NLAY * NREL * Dsz + 255) / 256, 256, 0, stream>>>(a_b, B_b, bc, bd);
    adj_pack_kernel<<<(NGr * NPAD * NPAD) / 256, 256, 0, stream>>>(graphs, adjT);
    init_pack_kernel<<<(int)(((size_t)NMIX * Bsz * Dsz * NPAD * Tsz) / 256), 256, 0, stream>>>(
        x0, x1, xsT, gbuf);

    for (int l = 0; l < NLAY; ++l) {
        // 4 combos * 96 bt * 32 mtiles = 12288 waves / 8 waves-per-block
        agg_kernel<<<(4 * BT * 32) / 8, 256, 0, stream>>>(adjT, xsT, agg);
        // 2 mixes * 96 bt * 32 mtiles = 6144 waves / 8
        dense_kernel<<<(NMIX * BT * 32) / 8, 256, 0, stream>>>(agg, wcB, wdB, bc, bd, hbuf, l);
        layer_pack_kernel<<<(int)(((size_t)NMIX * BT * Dsz * NPAD) / 256), 256, 0, stream>>>(
            hbuf, xsT, gbuf, l);
    }

    summarize_kernel<<<(int)(((size_t)NMIX * OUTPERI + 255) / 256), 256, 0, stream>>>(
        gbuf, nbr, nbrw, out);
}
